// MultiHeadAttention_76794015252865
// MI455X (gfx1250) — matrix operand order repaired and checked by reading, not compiled, not run on hardware
//
#include <hip/hip_runtime.h>
#include <hip/hip_bf16.h>

// ---------------- CDNA5 WMMA types ----------------
typedef __attribute__((ext_vector_type(16))) __bf16 v16bf;
typedef __attribute__((ext_vector_type(8)))  float  v8f;

__device__ __forceinline__ v8f wmma_bf16(v16bf a, v16bf b, v8f c) {
  // D = A(16x32 bf16) * B(32x16 bf16) + C(16x16 f32)
  return __builtin_amdgcn_wmma_f32_16x16x32_bf16(false, a, false, b,
                                                 (short)0, c, false, false);
}

// B-fragment (32x16): lane l holds 16 contiguous bf16 (K = (l<16?0:16)+0..15)
__device__ __forceinline__ v16bf ld_frag_contig(const __bf16* p) {
  union { v16bf v; uint4 q[2]; } u;
  u.q[0] = *(const uint4*)(p);
  u.q[1] = *(const uint4*)(p + 8);
  return u.v;
}
// A-fragment (16x32): lane l holds two separated 8-bf16 chunks
__device__ __forceinline__ v16bf ld_frag_split(const __bf16* p0, const __bf16* p1) {
  union { v16bf v; uint4 q[2]; } u;
  u.q[0] = *(const uint4*)(p0);
  u.q[1] = *(const uint4*)(p1);
  return u.v;
}

// ---- CDNA5 async global->LDS DMA (ASYNCcnt path, no VGPR round-trip) ----
__device__ __forceinline__ void async_copy16(unsigned lds_addr, const void* gaddr) {
  asm volatile("global_load_async_to_lds_b128 %0, %1, off"
               :: "v"(lds_addr), "v"((unsigned long long)(size_t)gaddr)
               : "memory");
}
__device__ __forceinline__ void wait_async0() {
  asm volatile("s_wait_asynccnt 0x0" ::: "memory");
}
__device__ __forceinline__ unsigned lds_off(const void* p) {
  return (unsigned)(size_t)p;   // flat LDS addr[31:0] == LDS byte offset
}

// Problem constants
#define NB   4
#define NH   8
#define LSEQ 2048
#define DIM  512
#define DK   64

// ---------------- W transpose+convert: W[K,N] f32 -> Wt[N,K] bf16 ----------
__global__ __launch_bounds__(256) void k_transpose_w(const float* __restrict__ W,
                                                     __bf16* __restrict__ Wt) {
  __shared__ float t[64][65];
  const int c0 = blockIdx.x * 64, r0 = blockIdx.y * 64;
  const int tc = threadIdx.x & 63, tr = threadIdx.x >> 6;
#pragma unroll
  for (int i = 0; i < 16; ++i) {
    const int rr = tr + i * 4;
    t[rr][tc] = W[(size_t)(r0 + rr) * DIM + c0 + tc];
  }
  __syncthreads();
#pragma unroll
  for (int i = 0; i < 16; ++i) {
    const int cc = tr + i * 4;
    Wt[(size_t)(c0 + cc) * DIM + r0 + tc] = (__bf16)t[tc][cc];
  }
}

// ---------------- Projection GEMM: P[8192,512] = X[8192,512] * W ----------
__global__ __launch_bounds__(256) void k_proj_gemm(const float* __restrict__ X,
                                                   const __bf16* __restrict__ Wt,
                                                   float* __restrict__ P) {
  const int lane = threadIdx.x & 31;
  const int wid  = threadIdx.x >> 5;
  const int tile = blockIdx.x * 8 + wid;   // 4096 waves
  const int m0 = (tile >> 3) * 16;         // 512 M-tiles
  const int n0 = (tile & 7) * 64;          // 8 N-blocks of 64
  const int half = lane >> 4;
  const int r = lane & 15;

  v8f acc[4] = {v8f{0}, v8f{0}, v8f{0}, v8f{0}};
  const float* xrow = X + (size_t)(m0 + r) * DIM;

  for (int k0 = 0; k0 < DIM; k0 += 32) {
    float af[16];
    {
      const float4* p0 = (const float4*)(xrow + k0 + (half ? 16 : 0));
      *(float4*)&af[0] = p0[0];
      *(float4*)&af[4] = p0[1];
      const float4* p1 = (const float4*)(xrow + k0 + (half ? 24 : 8));
      *(float4*)&af[8]  = p1[0];
      *(float4*)&af[12] = p1[1];
    }
    v16bf a;
#pragma unroll
    for (int i = 0; i < 16; ++i) a[i] = (__bf16)af[i];

#pragma unroll
    for (int d = 0; d < 4; ++d) {
      const __bf16* wrow = Wt + (size_t)(n0 + d * 16 + r) * DIM + k0 + (half ? 16 : 0);
      v16bf b = ld_frag_contig(wrow);
      acc[d] = wmma_bf16(a, b, acc[d]);
    }
  }
#pragma unroll
  for (int d = 0; d < 4; ++d)
#pragma unroll
    for (int i = 0; i < 8; ++i)
      P[(size_t)(m0 + i + half * 8) * DIM + n0 + d * 16 + r] = acc[d][i];
}

// ------------- L2-normalize 64-wide head rows, f32 -> bf16, fold scale -----
__global__ __launch_bounds__(256) void k_norm_rows(const float* __restrict__ P,
                                                   __bf16* __restrict__ Out,
                                                   float scale) {
  const int lane = threadIdx.x & 31;
  const int wid  = threadIdx.x >> 5;
  const int rrow = blockIdx.x * 8 + wid;      // 65536 head rows
  const int b = rrow >> 14;
  const int rem = rrow & 16383;
  const int l = rem >> 3;
  const int h = rem & 7;

  const float* src = P + ((size_t)(b * LSEQ + l) * DIM) + h * DK + lane * 2;
  const float x0 = src[0], x1 = src[1];
  float ss = x0 * x0 + x1 * x1;
#pragma unroll
  for (int m = 16; m >= 1; m >>= 1) ss += __shfl_xor(ss, m, 32);
  const float inv = scale / fmaxf(sqrtf(ss), 1e-12f);

  __bf16* dst = Out + ((size_t)((b * NH + h) * LSEQ + l) * DK) + lane * 2;
  dst[0] = (__bf16)(x0 * inv);
  dst[1] = (__bf16)(x1 * inv);
}

// ------------- V transpose: PV[B,L,512] f32 -> VT[bh,64,L] bf16 ------------
__global__ __launch_bounds__(256) void k_transpose_v(const float* __restrict__ PV,
                                                     __bf16* __restrict__ VT) {
  __shared__ float t[64][65];
  const int bh = blockIdx.z;
  const int b = bh >> 3, h = bh & 7;
  const int l0 = blockIdx.x * 64;
  const int tc = threadIdx.x & 63, tr = threadIdx.x >> 6;

  const float* src = PV + (size_t)b * LSEQ * DIM + (size_t)h * DK;
#pragma unroll
  for (int i = 0; i < 16; ++i) {
    const int rr = tr + i * 4;
    t[rr][tc] = src[(size_t)(l0 + rr) * DIM + tc];
  }
  __syncthreads();
  __bf16* dst = VT + (size_t)bh * DK * LSEQ;
#pragma unroll
  for (int i = 0; i < 16; ++i) {
    const int dc = tr + i * 4;
    dst[(size_t)dc * LSEQ + l0 + tc] = (__bf16)t[tc][dc];
  }
}

// ---------------- Fused attention ------------------------------------------
// 8 waves/block share one (b,h): K/V tiles async-DMA'd once per block into
// LDS (8x less L2 traffic), double-buffered: issue async copies for tile i+1,
// run WMMAs on tile i, s_wait_asynccnt + barrier.
// Pass 1: scores -> exp -> row-sums, fused O += E*V (E via per-wave LDS bounce)
// Pass 2: recompute scores from staged K, write w = exp(s)/sum
__global__ __launch_bounds__(256) void k_attention(const __bf16* __restrict__ QB,
                                                   const __bf16* __restrict__ KB,
                                                   const __bf16* __restrict__ VT,
                                                   float* __restrict__ O,
                                                   float* __restrict__ Wout) {
  __shared__ __attribute__((aligned(16))) __bf16 sk[2][32][64];    // K tiles, 4KB x2
  __shared__ __attribute__((aligned(16))) __bf16 sv[2][64][32];    // Vt tiles, 4KB x2
  __shared__ __attribute__((aligned(16))) __bf16 lds_e[8][16][40]; // per-wave E tile

  const int tid  = threadIdx.x;
  const int lane = tid & 31;
  const int wid  = tid >> 5;
  const int tile = blockIdx.x * 8 + wid;        // 4096 waves total
  const int bh = tile >> 7;                     // all 8 waves share bh
  const int m0 = (tile & 127) << 4;
  const int half = lane >> 4;
  const int r = lane & 15;

  const __bf16* qbase = QB + ((size_t)bh * LSEQ + m0) * DK;
  const __bf16* kbase = KB + (size_t)bh * LSEQ * DK;
  const __bf16* vbase = VT + (size_t)bh * DK * LSEQ;
  float* obase = O + ((size_t)bh * LSEQ + m0) * DK;
  float* wbase = Wout + ((size_t)bh << 22) + (size_t)m0 * LSEQ;

  // staging coordinates: K tile is 4KB contiguous; V tile is 64 strided rows
  const int vd = tid >> 2;            // d row 0..63
  const int vc = (tid & 3) * 8;       // 8-elem chunk within 32
  const unsigned skL[2] = { lds_off(&sk[0][0][0]) + (unsigned)tid * 16u,
                            lds_off(&sk[1][0][0]) + (unsigned)tid * 16u };
  const unsigned svL[2] = { lds_off(&sv[0][0][0]) + (unsigned)(vd * 32 + vc) * 2u,
                            lds_off(&sv[1][0][0]) + (unsigned)(vd * 32 + vc) * 2u };
  const __bf16* kg = kbase + tid * 8;                      // this thread's 16B of K tile
  const __bf16* vg = vbase + (size_t)vd * LSEQ + vc;       // this thread's 16B of V tile

  // Q A-fragments, held in VGPRs for the whole kernel
  const __bf16* qrow = qbase + (size_t)r * DK;
  const v16bf qa0 = ld_frag_split(qrow + (half ? 16 : 0),      qrow + (half ? 24 : 8));
  const v16bf qa1 = ld_frag_split(qrow + 32 + (half ? 16 : 0), qrow + 32 + (half ? 24 : 8));

  v8f lacc = {0};
  v8f oacc[4] = {v8f{0}, v8f{0}, v8f{0}, v8f{0}};

  // prologue: async-stage tile k0=0 into buffer 0
  async_copy16(skL[0], kg);
  async_copy16(svL[0], vg);
  wait_async0();
  __syncthreads();

  // ---------------- pass 1 ----------------
  for (int k0 = 0; k0 < LSEQ; k0 += 32) {
    const int buf = (k0 >> 5) & 1;
    const bool has_next = (k0 + 32) < LSEQ;
    if (has_next) {  // fire DMA for next tile; overlaps with WMMAs below
      async_copy16(skL[buf ^ 1], kg + (size_t)(k0 + 32) * DK);
      async_copy16(svL[buf ^ 1], vg + (k0 + 32));
    }

#pragma unroll
    for (int t = 0; t < 2; ++t) {
      const __bf16* krow = &sk[buf][t * 16 + r][half ? 16 : 0];
      union { v16bf v; uint4 q[2]; } u0, u1;
      u0.q[0] = *(const uint4*)(krow);
      u0.q[1] = *(const uint4*)(krow + 8);
      u1.q[0] = *(const uint4*)(krow + 32);
      u1.q[1] = *(const uint4*)(krow + 40);
      v8f s = {0};
      s = wmma_bf16(qa0, u0.v, s);
      s = wmma_bf16(qa1, u1.v, s);
      // scores bounded to [-1/8,1/8] by cosine norm: exp needs no max shift
#pragma unroll
      for (int i = 0; i < 8; ++i) {
        const float e = __expf(s[i]);
        lacc[i] += e;
        lds_e[wid][i + half * 8][t * 16 + r] = (__bf16)e;  // C-layout -> row-major
      }
    }
    // reload E as bf16 A-fragment (same-wave DS ops are in-order)
    union { v16bf v; uint4 q[2]; } ue;
    ue.q[0] = *(const uint4*)&lds_e[wid][r][half ? 16 : 0];
    ue.q[1] = *(const uint4*)&lds_e[wid][r][half ? 24 : 8];
#pragma unroll
    for (int d = 0; d < 4; ++d) {
      const __bf16* vrow = &sv[buf][d * 16 + r][half ? 16 : 0];
      union { v16bf v; uint4 q[2]; } uv;
      uv.q[0] = *(const uint4*)(vrow);
      uv.q[1] = *(const uint4*)(vrow + 8);
      oacc[d] = wmma_bf16(ue.v, uv.v, oacc[d]);
    }

    wait_async0();       // next tile landed in LDS
    __syncthreads();
  }

  // row-sum reduction within each 16-lane half (halves hold M 0-7 / 8-15)
#pragma unroll
  for (int m = 8; m >= 1; m >>= 1)
#pragma unroll
    for (int i = 0; i < 8; ++i) lacc[i] += __shfl_xor(lacc[i], m, 32);

  v8f invl;
#pragma unroll
  for (int i = 0; i < 8; ++i) invl[i] = 1.0f / lacc[i];

  // async-restage K tile 0 for pass 2 (overlaps with reduction/O stores)
  async_copy16(skL[0], kg);

  // write O = (sum e*v) / l
#pragma unroll
  for (int d = 0; d < 4; ++d)
#pragma unroll
    for (int i = 0; i < 8; ++i)
      obase[(size_t)(i + half * 8) * DK + d * 16 + r] = oacc[d][i] * invl[i];

  wait_async0();
  __syncthreads();

  // ---------------- pass 2: recompute scores, write w ----------------
  for (int k0 = 0; k0 < LSEQ; k0 += 32) {
    const int buf = (k0 >> 5) & 1;
    const bool has_next = (k0 + 32) < LSEQ;
    if (has_next)
      async_copy16(skL[buf ^ 1], kg + (size_t)(k0 + 32) * DK);

#pragma unroll
    for (int t = 0; t < 2; ++t) {
      const int n0 = k0 + t * 16;
      const __bf16* krow = &sk[buf][t * 16 + r][half ? 16 : 0];
      union { v16bf v; uint4 q[2]; } u0, u1;
      u0.q[0] = *(const uint4*)(krow);
      u0.q[1] = *(const uint4*)(krow + 8);
      u1.q[0] = *(const uint4*)(krow + 32);
      u1.q[1] = *(const uint4*)(krow + 40);
      v8f s = {0};
      s = wmma_bf16(qa0, u0.v, s);
      s = wmma_bf16(qa1, u1.v, s);
#pragma unroll
      for (int i = 0; i < 8; ++i)
        wbase[(size_t)(i + half * 8) * LSEQ + n0 + r] = __expf(s[i]) * invl[i];
    }

    wait_async0();
    __syncthreads();
  }
}

// ---------------------------- host launcher --------------------------------
extern "C" void kernel_launch(void* const* d_in, const int* in_sizes, int n_in,
                              void* d_out, int out_size, void* d_ws, size_t ws_size,
                              hipStream_t stream) {
  (void)in_sizes; (void)n_in; (void)out_size; (void)ws_size;
  const float* q  = (const float*)d_in[0];
  const float* k  = (const float*)d_in[1];
  const float* v  = (const float*)d_in[2];
  const float* Wq = (const float*)d_in[3];
  const float* Wk = (const float*)d_in[4];
  const float* Wv = (const float*)d_in[5];
  // d_in[6] = k_mask: the reference discards masked_fill, so it is a no-op.

  char* ws = (char*)d_ws;
  size_t off = 0;
  auto carve = [&](size_t bytes) { char* p = ws + off; off = (off + bytes + 255) & ~(size_t)255; return p; };

  const size_t projB = (size_t)NB * LSEQ * DIM * sizeof(float);      // 16 MB
  const size_t wtB   = (size_t)DIM * DIM * sizeof(__bf16);           // 512 KB
  const size_t headB = (size_t)NB * NH * LSEQ * DK * sizeof(__bf16); // 8 MB

  float*  PQ  = (float*)carve(projB);
  float*  PK  = (float*)carve(projB);
  float*  PV  = (float*)carve(projB);
  __bf16* WTQ = (__bf16*)carve(wtB);
  __bf16* WTK = (__bf16*)carve(wtB);
  __bf16* WTV = (__bf16*)carve(wtB);
  __bf16* QBn = (__bf16*)carve(headB);
  __bf16* KBn = (__bf16*)carve(headB);
  __bf16* VTB = (__bf16*)carve(headB);

  float* Optr = (float*)d_out;                                  // [B,H,Lq,64]
  float* Wptr = Optr + (size_t)NB * NH * LSEQ * DK;             // [B*H,Lq,Lk]

  // 1. pre-transpose/convert weights
  k_transpose_w<<<dim3(8, 8), 256, 0, stream>>>(Wq, WTQ);
  k_transpose_w<<<dim3(8, 8), 256, 0, stream>>>(Wk, WTK);
  k_transpose_w<<<dim3(8, 8), 256, 0, stream>>>(Wv, WTV);
  // 2. projections (bf16 WMMA, fp32 accumulate)
  k_proj_gemm<<<512, 256, 0, stream>>>(q, WTQ, PQ);
  k_proj_gemm<<<512, 256, 0, stream>>>(k, WTK, PK);
  k_proj_gemm<<<512, 256, 0, stream>>>(v, WTV, PV);
  // 3. cosine normalize (fold 1/sqrt(dk) into q), convert to bf16 head layout
  k_norm_rows<<<8192, 256, 0, stream>>>(PQ, QBn, 0.125f);
  k_norm_rows<<<8192, 256, 0, stream>>>(PK, KBn, 1.0f);
  // 4. V -> [bh, d, L] bf16 for contiguous B-fragments
  k_transpose_v<<<dim3(32, 1, 32), 256, 0, stream>>>(PV, VTB);
  // 5. fused attention
  k_attention<<<512, 256, 0, stream>>>(QBn, KBn, VTB, Optr, Wptr);
}
